// LSTM_52089363366074
// MI455X (gfx1250) — compile-verified, matrix operand order checked by reading, not run on hardware
//
#include <hip/hip_runtime.h>

// ---------------------------------------------------------------------------
// 2-layer LSTM (B=32, T=1024, D=256, H=512) for gfx1250.
// Persistent-RNN design: parallel WMMA GEMMs for input projections,
// single-WGP persistent scan kernel with h in LDS, c in VGPRs,
// bf16 v_wmma_f32_16x16x32_bf16 matmuls with fp32 accumulation.
// Streaming xp tiles are prefetched one timestep ahead (global_prefetch_b8)
// to keep HBM latency off the serial dependency chain.
// ---------------------------------------------------------------------------

typedef __bf16 bf16;
typedef __attribute__((ext_vector_type(16))) __bf16 v16bf;
typedef __attribute__((ext_vector_type(8)))  __bf16 v8bf;
typedef __attribute__((ext_vector_type(8)))  float  v8f;
typedef __attribute__((ext_vector_type(4)))  float  v4f;

#define T_STEPS 1024
#define BATCH   32
#define DIM     256
#define HID     512
#define G4      2048          // 4*H
#define NT_G    128           // G4/16 column tiles
#define KT_H    16            // HID/32 k-tiles
#define KT_D    8             // DIM/32 k-tiles

static __device__ __forceinline__ float sigmoidf_(float x) {
    return 1.0f / (1.0f + __expf(-x));
}

// --- fragment loaders ------------------------------------------------------
// A-fragment (16x32 bf16, ISA 7.12.2): lane L holds row m=L%16; with half=L/16
// the 16 elements are K = half*8 + {0..7} and K = 16 + half*8 + {0..7}.

static __device__ __forceinline__ v16bf load_afrag_f32(const float* __restrict__ base,
                                                       long rstride, int lane) {
    const int m = lane & 15, half = lane >> 4;
    const float* p = base + (size_t)m * rstride + half * 8;
    v4f q0 = *(const v4f*)(p);
    v4f q1 = *(const v4f*)(p + 4);
    v4f q2 = *(const v4f*)(p + 16);
    v4f q3 = *(const v4f*)(p + 20);
    v16bf a;
#pragma unroll
    for (int j = 0; j < 4; ++j) {
        a[j]      = (__bf16)q0[j];
        a[j + 4]  = (__bf16)q1[j];
        a[j + 8]  = (__bf16)q2[j];
        a[j + 12] = (__bf16)q3[j];
    }
    return a;
}

static __device__ __forceinline__ v16bf load_afrag_lds(const __bf16* h, int mt,
                                                       int kt, int lane) {
    const int m = (lane & 15) + mt * 16, half = lane >> 4;
    const __bf16* p = h + m * HID + kt * 32 + half * 8;
    union { v16bf v; struct { v8bf lo, hi; } s; } u;
    u.s.lo = *(const v8bf*)(p);        // ds_load_b128
    u.s.hi = *(const v8bf*)(p + 16);   // ds_load_b128
    return u.v;
}

// B-fragments are pre-packed so each lane reads its 16 bf16 contiguously.
static __device__ __forceinline__ v16bf load_bfrag(const bf16* __restrict__ wp,
                                                   int nt, int kt, int KT, int lane) {
    const bf16* p = wp + (((size_t)nt * KT + kt) * 32 + lane) * 16;
    return *(const v16bf*)p;           // 2x global_load_b128
}

static __device__ __forceinline__ v8f wmma_bf16(v16bf a, v16bf b, v8f c) {
    return __builtin_amdgcn_wmma_f32_16x16x32_bf16(false, a, false, b,
                                                   (short)0, c, false, false);
}

// --- weight packing: W [N=2048 rows][K cols] fp32 -> bf16 B-fragments ------
__global__ void pack_w(const float* __restrict__ W, bf16* __restrict__ out,
                       int K, int KT, int total_frags) {
    int id   = blockIdx.x * blockDim.x + threadIdx.x;
    int lane = id & 31;
    int fi   = id >> 5;
    if (fi >= total_frags) return;
    int kt = fi % KT;
    int nt = fi / KT;
    const float* src = W + (size_t)(nt * 16 + (lane & 15)) * K
                         + kt * 32 + (lane >> 4) * 8;
    v4f q0 = *(const v4f*)(src);
    v4f q1 = *(const v4f*)(src + 4);
    v4f q2 = *(const v4f*)(src + 16);
    v4f q3 = *(const v4f*)(src + 20);
    v16bf a;
#pragma unroll
    for (int j = 0; j < 4; ++j) {
        a[j]      = (__bf16)q0[j];
        a[j + 4]  = (__bf16)q1[j];
        a[j + 8]  = (__bf16)q2[j];
        a[j + 12] = (__bf16)q3[j];
    }
    *(v16bf*)(out + ((size_t)fi * 32 + lane) * 16) = a;
}

// --- parallel input projection: xp = src @ W^T, C-fragment-layout fp32 -----
// One wave per 16x16 output tile; grid covers T*128*2 tiles, whole chip busy.
__global__ __launch_bounds__(256)
void xp_gemm(const float* __restrict__ src, long tstr, long bstr, int KT,
             const bf16* __restrict__ wp, float* __restrict__ xp) {
    const int lane = threadIdx.x & 31;
    const int wid  = (blockIdx.x << 3) + (threadIdx.x >> 5);
    const int t    = wid >> 8;
    const int rem  = wid & 255;
    const int nt   = rem >> 1;
    const int mt   = rem & 1;
    v8f acc = {};
    const float* abase = src + (size_t)t * tstr + (size_t)(mt * 16) * bstr;
    for (int kt = 0; kt < KT; ++kt) {
        v16bf a = load_afrag_f32(abase + kt * 32, bstr, lane);
        v16bf b = load_bfrag(wp, nt, kt, KT, lane);
        acc = wmma_bf16(a, b, acc);
    }
    *(v8f*)(xp + ((size_t)(t * 256 + rem)) * 256 + lane * 8) = acc;
}

// --- persistent scan: one workgroup (16 waves), h in LDS, c in VGPRs -------
__global__ __launch_bounds__(512)
void lstm_rec(const float* __restrict__ xsrc, long x_tstr, long x_bstr, int KTx,
              const bf16* __restrict__ wih_p,
              const bf16* __restrict__ whh_p,
              const float* __restrict__ bias,
              const float* __restrict__ xp,
              float* __restrict__ hs_out, long h_tstr, long h_bstr,
              float* __restrict__ hn, float* __restrict__ cn) {
    __shared__ __bf16 hlds[2][BATCH * HID];   // 64 KB double-buffered h (bf16)

    const int lane = threadIdx.x & 31;
    const int w    = threadIdx.x >> 5;        // wave 0..15, owns h-cols [32w,32w+32)
    const int n    = lane & 15, half = lane >> 4;

    for (int i = threadIdx.x; i < BATCH * HID; i += 512) hlds[0][i] = (__bf16)0.0f;

    v8f zero = {};
    v8f c[4];                                  // cell state: [cc*2+mt], lives in VGPRs
#pragma unroll
    for (int i = 0; i < 4; ++i) c[i] = zero;

    float bv[2][4];                            // per-column gate biases (t-invariant)
#pragma unroll
    for (int cc = 0; cc < 2; ++cc)
#pragma unroll
        for (int g = 0; g < 4; ++g)
            bv[cc][g] = bias[g * HID + (w * 2 + cc) * 16 + n];

    __syncthreads();

    for (int t = 0; t < T_STEPS; ++t) {
        const __bf16* hcur  = hlds[t & 1];
        __bf16*       hnext = hlds[(t + 1) & 1];
        const float*  xpt   = xp ? (xp + (size_t)t * (256 * 256) + lane * 8) : nullptr;

        // Prefetch next timestep's streaming xp tiles (256 MB buffer exceeds
        // L2, so pull the HBM lines this wave will need at t+1 off the
        // critical path now). global_prefetch_b8: no counters, fire and forget.
        if (xpt && t + 1 < T_STEPS) {
            const float* nxt = xpt + (256 * 256);
#pragma unroll
            for (int g = 0; g < 4; ++g)
#pragma unroll
                for (int cc = 0; cc < 2; ++cc)
#pragma unroll
                    for (int mt = 0; mt < 2; ++mt)
                        __builtin_prefetch(nxt + (size_t)((g * 32 + (w * 2 + cc)) * 2 + mt) * 256,
                                           0, 0);
        }

#pragma unroll
        for (int cc = 0; cc < 2; ++cc) {
            const int ct = w * 2 + cc;
            v8f acc[8];                        // [gate*2 + mt]
            // init with bias (+ precomputed input projection if available)
#pragma unroll
            for (int g = 0; g < 4; ++g) {
#pragma unroll
                for (int mt = 0; mt < 2; ++mt) {
                    v8f a;
                    if (xpt) {
                        a = *(const v8f*)(xpt + (size_t)((g * 32 + ct) * 2 + mt) * 256);
#pragma unroll
                        for (int r = 0; r < 8; ++r) a[r] += bv[cc][g];
                    } else {
#pragma unroll
                        for (int r = 0; r < 8; ++r) a[r] = bv[cc][g];
                    }
                    acc[g * 2 + mt] = a;
                }
            }
            // fused input projection (concat-K fallback path, KTx==0 in xp path)
            for (int kt = 0; kt < KTx; ++kt) {
                v16bf a0 = load_afrag_f32(xsrc + (size_t)t * x_tstr + kt * 32,
                                          x_bstr, lane);
                v16bf a1 = load_afrag_f32(xsrc + (size_t)t * x_tstr +
                                          (size_t)16 * x_bstr + kt * 32,
                                          x_bstr, lane);
#pragma unroll
                for (int g = 0; g < 4; ++g) {
                    v16bf bf = load_bfrag(wih_p, g * 32 + ct, kt, KTx, lane);
                    acc[g * 2 + 0] = wmma_bf16(a0, bf, acc[g * 2 + 0]);
                    acc[g * 2 + 1] = wmma_bf16(a1, bf, acc[g * 2 + 1]);
                }
            }
            // recurrent part: gates += h_{t-1} @ W_hh^T (A from LDS, B from L2)
#pragma unroll 4
            for (int kt = 0; kt < KT_H; ++kt) {
                v16bf a0 = load_afrag_lds(hcur, 0, kt, lane);
                v16bf a1 = load_afrag_lds(hcur, 1, kt, lane);
#pragma unroll
                for (int g = 0; g < 4; ++g) {
                    v16bf bf = load_bfrag(whh_p, g * 32 + ct, kt, KT_H, lane);
                    acc[g * 2 + 0] = wmma_bf16(a0, bf, acc[g * 2 + 0]);
                    acc[g * 2 + 1] = wmma_bf16(a1, bf, acc[g * 2 + 1]);
                }
            }
            // elementwise cell update; C-layout: entry r on lane -> (m=r+8*half, n)
            const int col = ct * 16 + n;
#pragma unroll
            for (int mt = 0; mt < 2; ++mt) {
                v8f cv = c[cc * 2 + mt];
#pragma unroll
                for (int r = 0; r < 8; ++r) {
                    float iv = sigmoidf_(acc[0 * 2 + mt][r]);
                    float fv = sigmoidf_(acc[1 * 2 + mt][r]);
                    float gv = tanhf(acc[2 * 2 + mt][r]);
                    float ov = sigmoidf_(acc[3 * 2 + mt][r]);
                    float cnew = fv * cv[r] + iv * gv;
                    cv[r] = cnew;
                    float hv = ov * tanhf(cnew);
                    const int b = mt * 16 + r + 8 * half;
                    hnext[b * HID + col] = (__bf16)hv;
                    hs_out[(size_t)t * h_tstr + (size_t)b * h_bstr + col] = hv;
                    if (t == T_STEPS - 1) {
                        hn[b * HID + col] = hv;
                        cn[b * HID + col] = cnew;
                    }
                }
                c[cc * 2 + mt] = cv;
            }
        }
        __syncthreads();   // one cheap workgroup barrier per timestep
    }
}

// ---------------------------------------------------------------------------
extern "C" void kernel_launch(void* const* d_in, const int* in_sizes, int n_in,
                              void* d_out, int out_size, void* d_ws, size_t ws_size,
                              hipStream_t stream) {
    (void)in_sizes; (void)n_in; (void)out_size;
    const float* x    = (const float*)d_in[0];
    const float* Wih0 = (const float*)d_in[1];
    const float* b0   = (const float*)d_in[2];
    const float* Whh0 = (const float*)d_in[3];
    const float* Wih1 = (const float*)d_in[4];
    const float* b1   = (const float*)d_in[5];
    const float* Whh1 = (const float*)d_in[6];
    float* out = (float*)d_out;

    char* ws = (char*)d_ws;
    const size_t SZ_WIH0 = (size_t)NT_G * KT_D * 32 * 16 * sizeof(bf16); // 1 MB
    const size_t SZ_W    = (size_t)NT_G * KT_H * 32 * 16 * sizeof(bf16); // 2 MB
    bf16* wih0_p = (bf16*)(ws);
    bf16* whh0_p = (bf16*)(ws + SZ_WIH0);
    bf16* wih1_p = (bf16*)(ws + SZ_WIH0 + SZ_W);
    bf16* whh1_p = (bf16*)(ws + SZ_WIH0 + 2 * SZ_W);
    float* hs0   = (float*)(ws + SZ_WIH0 + 3 * SZ_W);
    const size_t SZ_HS0 = (size_t)T_STEPS * BATCH * HID * sizeof(float);  // 64 MB
    float* xp    = (float*)(ws + SZ_WIH0 + 3 * SZ_W + SZ_HS0);
    const size_t SZ_XP  = (size_t)T_STEPS * BATCH * G4 * sizeof(float);   // 256 MB
    const bool use_xp = ws_size >= (SZ_WIH0 + 3 * SZ_W + SZ_HS0 + SZ_XP);

    // one-time weight packing into WMMA B-fragment layout (bf16)
    const int frags_d = NT_G * KT_D;   // 1024 frags
    const int frags_h = NT_G * KT_H;   // 2048 frags
    pack_w<<<(frags_d * 32) / 256, 256, 0, stream>>>(Wih0, wih0_p, DIM, KT_D, frags_d);
    pack_w<<<(frags_h * 32) / 256, 256, 0, stream>>>(Whh0, whh0_p, HID, KT_H, frags_h);
    pack_w<<<(frags_h * 32) / 256, 256, 0, stream>>>(Wih1, wih1_p, HID, KT_H, frags_h);
    pack_w<<<(frags_h * 32) / 256, 256, 0, stream>>>(Whh1, whh1_p, HID, KT_H, frags_h);

    float* hn0 = out + (size_t)BATCH * T_STEPS * HID;   // output is [B,T,H]
    float* hn1 = hn0 + BATCH * HID;
    float* cn0 = hn1 + BATCH * HID;
    float* cn1 = cn0 + BATCH * HID;

    const long X_TSTR   = DIM;                   // x[b][t][d]
    const long X_BSTR   = (long)T_STEPS * DIM;
    const long HS0_TSTR = (long)BATCH * HID;     // hs0[t][b][h]
    const long HS0_BSTR = HID;
    const long OUT_TSTR = HID;                   // out[b][t][h]
    const long OUT_BSTR = (long)T_STEPS * HID;

    if (use_xp) {
        // parallel input projections across the whole chip, then serial scans
        xp_gemm<<<32768, 256, 0, stream>>>(x, X_TSTR, X_BSTR, KT_D, wih0_p, xp);
        lstm_rec<<<1, 512, 0, stream>>>(x, X_TSTR, X_BSTR, 0, wih0_p, whh0_p, b0, xp,
                                        hs0, HS0_TSTR, HS0_BSTR, hn0, cn0);
        xp_gemm<<<32768, 256, 0, stream>>>(hs0, HS0_TSTR, HS0_BSTR, KT_H, wih1_p, xp);
        lstm_rec<<<1, 512, 0, stream>>>(hs0, HS0_TSTR, HS0_BSTR, 0, wih1_p, whh1_p, b1, xp,
                                        out, OUT_TSTR, OUT_BSTR, hn1, cn1);
    } else {
        // workspace-lean fallback: fuse input projection into the scan (concat-K)
        lstm_rec<<<1, 512, 0, stream>>>(x, X_TSTR, X_BSTR, KT_D, wih0_p, whh0_p, b0,
                                        nullptr, hs0, HS0_TSTR, HS0_BSTR, hn0, cn0);
        lstm_rec<<<1, 512, 0, stream>>>(hs0, HS0_TSTR, HS0_BSTR, KT_H, wih1_p, whh1_p, b1,
                                        nullptr, out, OUT_TSTR, OUT_BSTR, hn1, cn1);
    }
}
